// SNN_33706903339705
// MI455X (gfx1250) — matrix operand order repaired and checked by reading, not compiled, not run on hardware
//
#include <hip/hip_runtime.h>

// ---------------------------------------------------------------------------
// DAG-MLP forward for MI455X (gfx1250), wave32, v_wmma_f32_16x16x32_bf16.
//
//   1) k_maskconv     : Whm = bf16(Wh * mask)  (HBM-bound once; 94MB result is
//                       L2-resident for all GEMMs: 192MB global L2)
//   2) k_pad_bf16     : x, Win -> bf16, K padded 784->800
//   3) k_gemm_partial : per predecessor-block j, partial_j = A_j @ B_j^T
//                       grid (64 N-stripes x j-blocks) -> up to 4608 waves,
//                       32 K-chunks/wave, unroll 4 for load/WMMA overlap
//   4) k_reduce       : acts[i] = bf16(relu(sum_j partial_j + bias))
//                       fixed-order sum => deterministic (no float atomics)
//   5) k_final        : 128x10 output layer (tiny, scalar fp32)
// ---------------------------------------------------------------------------

typedef __attribute__((ext_vector_type(16))) __bf16 v16bf;
typedef __attribute__((ext_vector_type(8)))  __bf16 v8bf;
typedef __attribute__((ext_vector_type(4)))  __bf16 v4bf;
typedef __attribute__((ext_vector_type(8)))  float  v8f;

#define WIDTH   1024
#define BATCH   128
#define DIN     784
#define DPAD    800     // 784 padded to a multiple of 32
#define NLAYERS 10
#define NPAIRS  45
#define NCLS    10

// --- 1) masked weight convert: Whm[e] = bf16(Wh[e] * mask[e]) ---------------
__global__ __launch_bounds__(256) void k_maskconv(const float* __restrict__ w,
                                                  const unsigned char* __restrict__ m,
                                                  __bf16* __restrict__ out,
                                                  long long n8) {
    long long i = (long long)blockIdx.x * 256 + threadIdx.x;
    if (i >= n8) return;
    long long base = i * 8;
    const float4* w4 = (const float4*)(w + base);
    float4 a = w4[0], b = w4[1];
    unsigned long long mb = *(const unsigned long long*)(m + base);
    float vals[8] = {a.x, a.y, a.z, a.w, b.x, b.y, b.z, b.w};
    v8bf o;
#pragma unroll
    for (int e = 0; e < 8; ++e) {
        unsigned char keep = (unsigned char)(mb >> (8 * e));
        o[e] = keep ? (__bf16)vals[e] : (__bf16)0.0f;
    }
    *(v8bf*)(out + base) = o;
}

// --- 2) fp32 -> bf16 with zero pad of the K dimension -----------------------
__global__ __launch_bounds__(256) void k_pad_bf16(const float* __restrict__ src,
                                                  __bf16* __restrict__ dst,
                                                  int rows, int scols, int dcols) {
    long long i = (long long)blockIdx.x * 256 + threadIdx.x;
    long long n = (long long)rows * dcols;
    if (i >= n) return;
    int r = (int)(i / dcols);
    int c = (int)(i % dcols);
    float v = (c < scols) ? src[(long long)r * scols + c] : 0.0f;
    dst[i] = (__bf16)v;
}

// --- 3) split-K GEMM partial: partial[jb] = A_jb @ B_jb^T -------------------
// A_jb : [128 x kPer] bf16 row-major (lda), block stride aStride
// B_jb : [1024 x kPer] bf16 row-major (ldb) == B^T of the GEMM, stride bStride
// Grid: (64, nblocks) x 256 thr; wave = M tile, blockIdx.x = N stripe,
// blockIdx.y = jb. Each wave: 16x16 tile over kPer (<=32 chunks).
__global__ __launch_bounds__(256) void k_gemm_partial(
        const __bf16* __restrict__ Abase, long long aStride, int lda,
        const __bf16* __restrict__ Bbase, long long bStride, int ldb,
        int kPer, float* __restrict__ partial) {
    int wave = threadIdx.x >> 5;
    int lane = threadIdx.x & 31;
    int jb   = blockIdx.y;
    int m0   = wave * 16;            // 8 M tiles (batch)
    int n0   = blockIdx.x * 16;      // 64 N stripes (features)
    int r16  = lane & 15;            // A row / B column within tile
    int grp  = lane >> 4;            // lane half selects K sub-ranges

    const __bf16* A  = Abase + (long long)jb * aStride
                     + (long long)(m0 + r16) * lda + grp * 8;
    const __bf16* Bp = Bbase + (long long)jb * bStride
                     + (long long)(n0 + r16) * ldb + grp * 8;

    v8f acc = {};
#pragma unroll 4
    for (int k = 0; k < kPer; k += 32) {
        __builtin_prefetch(Bp + k + 512, 0, 0);   // global_prefetch on weights
        v8bf a0 = *(const v8bf*)(A + k);          // A: K = k+grp*8    .. +7
        v8bf a1 = *(const v8bf*)(A + k + 16);     // A: K = k+grp*8+16 .. +23
        v8bf b0 = *(const v8bf*)(Bp + k);         // B: K = k+grp*8    .. +7
        v8bf b1 = *(const v8bf*)(Bp + k + 16);    // B: K = k+grp*8+16 .. +23
        v16bf av, bv;
#pragma unroll
        for (int e = 0; e < 8; ++e) {
            av[e] = a0[e]; av[e + 8] = a1[e];
            bv[e] = b0[e]; bv[e + 8] = b1[e];
        }
        acc = __builtin_amdgcn_wmma_f32_16x16x32_bf16(
                  false, av, false, bv, (short)0, acc, false, false);
    }
    // C/D layout: VGPR r holds (M = m0 + r + 8*grp, N = n0 + (lane&15))
    float* outp = partial + (long long)jb * (BATCH * WIDTH);
    int col = n0 + r16;
#pragma unroll
    for (int r = 0; r < 8; ++r) {
        int row = m0 + r + grp * 8;
        outp[(long long)row * WIDTH + col] = acc[r];
    }
}

// --- 4) deterministic reduction: dst = bf16(relu(sum_j partial_j + bias)) ---
__global__ __launch_bounds__(256) void k_reduce(const float* __restrict__ partial,
                                                int nblocks,
                                                const float* __restrict__ bias,
                                                __bf16* __restrict__ dst) {
    int i4 = blockIdx.x * 256 + threadIdx.x;           // vec4 index
    if (i4 >= BATCH * WIDTH / 4) return;
    long long base = (long long)i4 * 4;
    float4 s = *(const float4*)(partial + base);
    for (int j = 1; j < nblocks; ++j) {                // fixed order => determinism
        float4 p = *(const float4*)(partial + (long long)j * BATCH * WIDTH + base);
        s.x += p.x; s.y += p.y; s.z += p.z; s.w += p.w;
    }
    int col = (int)(base % WIDTH);
    s.x += bias[col];     s.y += bias[col + 1];
    s.z += bias[col + 2]; s.w += bias[col + 3];
    v4bf o;
    o[0] = (__bf16)(s.x > 0.0f ? s.x : 0.0f);
    o[1] = (__bf16)(s.y > 0.0f ? s.y : 0.0f);
    o[2] = (__bf16)(s.z > 0.0f ? s.z : 0.0f);
    o[3] = (__bf16)(s.w > 0.0f ? s.w : 0.0f);
    *(v4bf*)(dst + base) = o;
}

// --- 5) output layer: out[b,c] = acts9[b,:] . Wout[c,:] + bout[c] -----------
__global__ __launch_bounds__(128) void k_final(const __bf16* __restrict__ act,
                                               const float* __restrict__ Wout,
                                               const float* __restrict__ bout,
                                               float* __restrict__ out) {
    int i = blockIdx.x * 128 + threadIdx.x;
    if (i >= BATCH * NCLS) return;
    int b = i / NCLS, c = i % NCLS;
    float s = bout[c];
    const __bf16* ap = act + (long long)b * WIDTH;
    const float*  wp = Wout + (long long)c * WIDTH;
    for (int k = 0; k < WIDTH; ++k) s += (float)ap[k] * wp[k];
    out[i] = s;
}

// ---------------------------------------------------------------------------
extern "C" void kernel_launch(void* const* d_in, const int* in_sizes, int n_in,
                              void* d_out, int out_size, void* d_ws, size_t ws_size,
                              hipStream_t stream) {
    (void)in_sizes; (void)n_in; (void)out_size; (void)ws_size;
    const float*         x     = (const float*)d_in[0];
    const float*         Win   = (const float*)d_in[1];
    const float*         b_in  = (const float*)d_in[2];
    const float*         Wh    = (const float*)d_in[3];
    const float*         bh    = (const float*)d_in[4];
    const unsigned char* masks = (const unsigned char*)d_in[5];   // jax bool = u8
    const float*         Wout  = (const float*)d_in[6];
    const float*         bout  = (const float*)d_in[7];
    float*               out   = (float*)d_out;

    // Workspace layout (all 16B-aligned):
    char*   ws      = (char*)d_ws;
    __bf16* Whm     = (__bf16*)ws;                                        // 45*1024*1024 bf16
    __bf16* WinPad  = (__bf16*)(ws + (size_t)NPAIRS * WIDTH * WIDTH * 2); // 1024*800 bf16
    __bf16* xPad    = (__bf16*)((char*)WinPad + (size_t)WIDTH * DPAD * 2);// 128*800 bf16
    __bf16* acts    = (__bf16*)((char*)xPad + (size_t)BATCH * DPAD * 2);  // 10*128*1024 bf16
    float*  partial = (float*)((char*)acts + (size_t)NLAYERS * BATCH * WIDTH * 2); // 9*128*1024 f32

    // 1) masked weight convert (HBM-bound; result becomes L2-resident)
    long long n8 = (long long)NPAIRS * WIDTH * WIDTH / 8;
    k_maskconv<<<(int)((n8 + 255) / 256), 256, 0, stream>>>(Wh, masks, Whm, n8);

    // 2) pad+convert input-layer operands
    k_pad_bf16<<<(WIDTH * DPAD + 255) / 256, 256, 0, stream>>>(Win, WinPad, WIDTH, DIN, DPAD);
    k_pad_bf16<<<(BATCH * DPAD + 255) / 256, 256, 0, stream>>>(x,   xPad,   BATCH, DIN, DPAD);

    const int redBlocks = (BATCH * WIDTH / 4 + 255) / 256;   // 128

    // 3) layer 0: acts[0] = relu(x @ Win^T + b_in)
    k_gemm_partial<<<dim3(64, 1), 256, 0, stream>>>(xPad, 0LL, DPAD,
                                                    WinPad, 0LL, DPAD,
                                                    DPAD, partial);
    k_reduce<<<redBlocks, 256, 0, stream>>>(partial, 1, b_in, acts);

    // layers 1..9: acts[i] = relu( sum_j acts[j] @ (Wh*mask)[off+j]^T + bh[i-1] )
    int off = 0;
    for (int i = 1; i < NLAYERS; ++i) {
        k_gemm_partial<<<dim3(64, i), 256, 0, stream>>>(
            acts, (long long)BATCH * WIDTH, WIDTH,
            Whm + (size_t)off * WIDTH * WIDTH, (long long)WIDTH * WIDTH, WIDTH,
            WIDTH, partial);
        k_reduce<<<redBlocks, 256, 0, stream>>>(partial, i,
                                                bh + (size_t)(i - 1) * WIDTH,
                                                acts + (size_t)i * BATCH * WIDTH);
        off += i;
    }

    // 4) output layer
    k_final<<<(BATCH * NCLS + 127) / 128, 128, 0, stream>>>(
        acts + (size_t)9 * BATCH * WIDTH, Wout, bout, out);
}
